// MoELayer_9234179687043
// MI455X (gfx1250) — matrix operand order repaired and checked by reading, not compiled, not run on hardware
//
#include <hip/hip_runtime.h>
#include <hip/hip_bf16.h>
#include <stdint.h>

// Problem constants (match reference)
#define T_TOK  8192   // B*S
#define DDIM   512
#define FDIM   1024
#define NEXP   8
#define TOPK   2

#define MT     64     // tokens per block
#define FC     64     // F-chunk
#define TPB    512    // 16 waves of 32

// LDS strides (bf16 halves). Rows are 8B-aligned and stride == 2 (mod 4) dwords:
// 32-lane row walks hit 32 distinct even banks (+1 covers odd) -> conflict-free.
#define XS_STRIDE  516          // 1032 B = 129 * 8
#define W1S_STRIDE 516          // W1 chunk [col][k]
#define W2S_STRIDE 68           // 136 B = 17 * 8, W2 chunk [col][k]
#define HS_STRIDE  68
#define WS_HALVES  (DDIM * W2S_STRIDE)   // 34816 >= 64*516=33024

typedef __attribute__((ext_vector_type(16))) __bf16 v16bf;
typedef __attribute__((ext_vector_type(8)))  float  v8f;

// ---- fragment loaders (ISA 7.12.2 layouts), 8B-granular ------------------
__device__ inline v16bf load_b_frag(const __bf16* p) {  // K k0..k0+7, k0+16..k0+23
  union { v16bf v; uint64_t u[4]; } r;
  const uint64_t* q = (const uint64_t*)p;
  r.u[0] = q[0]; r.u[1] = q[1]; r.u[2] = q[4]; r.u[3] = q[5];
  return r.v;
}
__device__ inline v16bf load_a_frag(const __bf16* p) {  // K k0..k0+7, k0+16..k0+23
  union { v16bf v; uint64_t u[4]; } r;
  const uint64_t* q = (const uint64_t*)p;
  r.u[0] = q[0]; r.u[1] = q[1]; r.u[2] = q[4]; r.u[3] = q[5];
  return r.v;
}

__device__ inline uint16_t bf16_bits(float f) {
  __bf16 b = (__bf16)f;
  union { __bf16 b; uint16_t s; } u; u.b = b;
  return u.s;
}

__device__ inline float gelu_tanh(float x) {
  const float c0 = 0.7978845608028654f;   // sqrt(2/pi)
  float x3 = x * x * x;
  float t  = tanhf(c0 * (x + 0.044715f * x3));
  return 0.5f * x * (1.0f + t);
}

// ---- kernel 0: transpose + convert weights to bf16 in workspace ----------
// src [E][R][C] fp32 -> dst [E][C][R] bf16
__global__ __launch_bounds__(256) void moe_wtrans_kernel(
    const float* __restrict__ src, __bf16* __restrict__ dst, int R, int C) {
  __shared__ float tile[32][33];
  const size_t eoff = (size_t)blockIdx.z * R * C;
  const float*  s = src + eoff;
  __bf16*       d = dst + eoff;
  const int c0 = blockIdx.x * 32;
  const int r0 = blockIdx.y * 32;
  const int tx = threadIdx.x & 31;
  const int ty = threadIdx.x >> 5;          // 0..7
#pragma unroll
  for (int i = 0; i < 4; ++i)
    tile[ty + 8 * i][tx] = s[(size_t)(r0 + ty + 8 * i) * C + c0 + tx];
  __syncthreads();
#pragma unroll
  for (int i = 0; i < 4; ++i)
    d[(size_t)(c0 + ty + 8 * i) * R + r0 + tx] = (__bf16)tile[tx][ty + 8 * i];
}

// ---- kernel 1: combine weights cw[E][T] ----------------------------------
__global__ __launch_bounds__(256) void moe_cw_kernel(
    const int* __restrict__ idx, const float* __restrict__ w,
    float* __restrict__ cw) {
  int t = blockIdx.x * blockDim.x + threadIdx.x;
  if (t >= T_TOK) return;
  float acc[NEXP];
#pragma unroll
  for (int e = 0; e < NEXP; ++e) acc[e] = 0.0f;
#pragma unroll
  for (int k = 0; k < TOPK; ++k) {
    int e = idx[t * TOPK + k];
    acc[e] += w[t * TOPK + k];
  }
#pragma unroll
  for (int e = 0; e < NEXP; ++e) cw[e * T_TOK + t] = acc[e];
}

// ---- kernel 2: fused MoE FFN over all experts ----------------------------
// W1t: [E][F][D] bf16 (W1 transposed), W2t: [E][D][F] bf16 (W2 transposed)
__global__ __launch_bounds__(TPB) void moe_ffn_kernel(
    const float*  __restrict__ x,
    const __bf16* __restrict__ W1t, const float* __restrict__ b1,
    const __bf16* __restrict__ W2t, const float* __restrict__ b2,
    const float*  __restrict__ cw, float* __restrict__ out) {
  __shared__ __align__(16) float  cws[NEXP * MT];       // 2 KB
  __shared__ __align__(16) __bf16 Xs[MT * XS_STRIDE];   // 66 KB  [m][k]
  __shared__ __align__(16) __bf16 Ws[WS_HALVES];        // 69.6 KB [col][k]
  __shared__ __align__(16) __bf16 Hs[MT * HS_STRIDE];   // 8.7 KB [m][f]

  const int tid  = threadIdx.x;
  const int tok0 = blockIdx.x * MT;

  // ---- stage X tile (fp32 -> bf16, pairwise) and combine weights
  {
    const float2* x2 = (const float2*)(x + (size_t)tok0 * DDIM);
    uint32_t* xd = (uint32_t*)Xs;
#pragma unroll 4
    for (int i = tid; i < MT * (DDIM / 2); i += TPB) {   // 16384 pairs
      int m  = i >> 8;             // /256
      int k2 = i & 255;
      float2 f = x2[m * 256 + k2];
      uint32_t p = (uint32_t)bf16_bits(f.x) | ((uint32_t)bf16_bits(f.y) << 16);
      xd[m * (XS_STRIDE / 2) + k2] = p;
    }
  }
  {
    int e = tid >> 6;          // 0..7
    int t = tid & 63;          // 0..63  (E*MT == TPB exactly)
    cws[e * MT + t] = cw[(size_t)e * T_TOK + tok0 + t];
  }
  __syncthreads();

  const int wv    = tid >> 5;         // wave 0..15
  const int lane  = tid & 31;
  const int lhalf = lane >> 4;        // 0/1 half-wave
  const int lmod  = lane & 15;
  const int m_grp = wv & 3;           // 16-token row group, 0..3
  const int n_grp = wv >> 2;          // 0..3

  v8f acc[8];
#pragma unroll
  for (int j = 0; j < 8; ++j) acc[j] = (v8f){0.f,0.f,0.f,0.f,0.f,0.f,0.f,0.f};

  const __bf16* xrow = &Xs[(m_grp * 16 + lmod) * XS_STRIDE];
  const __bf16* wcol = &Ws[(n_grp * 16 + lmod) * W1S_STRIDE];
  const __bf16* hrow = &Hs[(m_grp * 16 + lmod) * HS_STRIDE];

  for (int e = 0; e < NEXP; ++e) {
    const __bf16* W1te = W1t + (size_t)e * FDIM * DDIM;   // [F][D]
    const __bf16* W2te = W2t + (size_t)e * DDIM * FDIM;   // [D][F]
    float cwr[8];
#pragma unroll
    for (int r = 0; r < 8; ++r)
      cwr[r] = cws[e * MT + m_grp * 16 + r + 8 * lhalf];

    for (int fc = 0; fc < FDIM; fc += FC) {
      // ---- stage W1 chunk: 64 contiguous bf16 rows of W1t, 8B copies -----
      __syncthreads();   // previous GEMM2 done with Ws/Hs
      {
        const uint64_t* g = (const uint64_t*)(W1te + (size_t)fc * DDIM);
        uint64_t* wd = (uint64_t*)Ws;
#pragma unroll 4
        for (int i = tid; i < FC * (DDIM / 4); i += TPB) {   // 8192 qwords
          int k4 = i & 127;          // qword within 512-half row
          int c  = i >> 7;           // 0..63
          wd[c * (W1S_STRIDE / 4) + k4] = g[c * (DDIM / 4) + k4];
        }
      }
      __syncthreads();

      // ---- GEMM1 (pipelined): hC = X(m_grp) * W1[:, fc+n_grp*16..], K=512
      v8f hC = (v8f){0.f,0.f,0.f,0.f,0.f,0.f,0.f,0.f};
      {
        v16bf a = load_a_frag(xrow + 8 * lhalf);
        v16bf b = load_b_frag(wcol + 8 * lhalf);
#pragma unroll
        for (int ks = 0; ks < DDIM / 32; ++ks) {
          v16bf an, bn;
          if (ks < DDIM / 32 - 1) {
            an = load_a_frag(xrow + (ks + 1) * 32 + 8 * lhalf);
            bn = load_b_frag(wcol + (ks + 1) * 32 + 8 * lhalf);
          }
          hC = __builtin_amdgcn_wmma_f32_16x16x32_bf16(false, a, false, b,
                                                       (short)0, hC,
                                                       false, false);
          if (ks < DDIM / 32 - 1) { a = an; b = bn; }
        }
      }
      // epilogue: + b1, gelu, * cw(row)  -> Hs (bf16)
      float b1v = b1[(size_t)e * FDIM + fc + n_grp * 16 + lmod];
#pragma unroll
      for (int r = 0; r < 8; ++r) {
        int row = m_grp * 16 + r + 8 * lhalf;
        float v = hC[r] + b1v;
        v = gelu_tanh(v) * cwr[r];
        Hs[row * HS_STRIDE + n_grp * 16 + lmod] = (__bf16)v;
      }
      __syncthreads();   // GEMM1 done reading Ws; Hs written

      // ---- stage W2 chunk: Ws[n][kk] = W2t[n][fc+kk], 8B copies ----------
      {
        uint64_t* wd = (uint64_t*)Ws;
#pragma unroll 4
        for (int i = tid; i < DDIM * (FC / 4); i += TPB) {   // 8192 qwords
          int kk4 = i & 15;          // qword within 64-half run
          int n   = i >> 4;          // 0..511
          wd[n * (W2S_STRIDE / 4) + kk4] =
              ((const uint64_t*)(W2te + (size_t)n * FDIM + fc))[kk4];
        }
      }
      __syncthreads();

      // ---- GEMM2 (pipelined): acc(64x512) += Hs(64x64) * W2chunk(64x512) -
      {
        v16bf a2[2];
        a2[0] = load_a_frag(hrow + 8 * lhalf);
        a2[1] = load_a_frag(hrow + 32 + 8 * lhalf);
        const __bf16* c0 = &Ws[((n_grp * 8) * 16 + lmod) * W2S_STRIDE];
        v16bf bA = load_b_frag(c0 + 8 * lhalf);
        v16bf bB = load_b_frag(c0 + 32 + 8 * lhalf);
#pragma unroll
        for (int j = 0; j < 8; ++j) {
          v16bf nA, nB;
          if (j < 7) {
            const __bf16* cn = &Ws[((n_grp * 8 + j + 1) * 16 + lmod) * W2S_STRIDE];
            nA = load_b_frag(cn + 8 * lhalf);
            nB = load_b_frag(cn + 32 + 8 * lhalf);
          }
          acc[j] = __builtin_amdgcn_wmma_f32_16x16x32_bf16(false, a2[0], false, bA,
                                                           (short)0, acc[j],
                                                           false, false);
          acc[j] = __builtin_amdgcn_wmma_f32_16x16x32_bf16(false, a2[1], false, bB,
                                                           (short)0, acc[j],
                                                           false, false);
          if (j < 7) { bA = nA; bB = nB; }
        }
      }
    }
  }

  // ---- epilogue: add sum_e cw[e,row]*b2[e,n], write out --------------------
#pragma unroll
  for (int j = 0; j < 8; ++j) {
    int n = (n_grp * 8 + j) * 16 + lmod;
    float b2l[NEXP];
#pragma unroll
    for (int e = 0; e < NEXP; ++e) b2l[e] = b2[(size_t)e * DDIM + n];
#pragma unroll
    for (int r = 0; r < 8; ++r) {
      int row = m_grp * 16 + r + 8 * lhalf;
      float v = acc[j][r];
#pragma unroll
      for (int e = 0; e < NEXP; ++e) v += cws[e * MT + row] * b2l[e];
      out[(size_t)(tok0 + row) * DDIM + n] = v;
    }
  }
}

// ---- launcher -------------------------------------------------------------
extern "C" void kernel_launch(void* const* d_in, const int* in_sizes, int n_in,
                              void* d_out, int out_size, void* d_ws, size_t ws_size,
                              hipStream_t stream) {
  const float* hidden = (const float*)d_in[0];
  const int*   idx    = (const int*)  d_in[1];
  const float* tkw    = (const float*)d_in[2];
  const float* W1     = (const float*)d_in[3];
  const float* b1     = (const float*)d_in[4];
  const float* W2     = (const float*)d_in[5];
  const float* b2     = (const float*)d_in[6];
  float* out = (float*)d_out;

  // workspace layout: cw | W1t (bf16) | W2t (bf16)  (~17 MB total)
  char* ws = (char*)d_ws;
  float*  cwp = (float*)ws;                                   // E*T*4 = 256 KB
  __bf16* W1t = (__bf16*)(ws + (size_t)NEXP * T_TOK * 4);     // E*F*D*2 = 8 MB
  __bf16* W2t = W1t + (size_t)NEXP * FDIM * DDIM;             // E*D*F*2 = 8 MB

  moe_cw_kernel<<<T_TOK / 256, 256, 0, stream>>>(idx, tkw, cwp);
  {
    dim3 g1(FDIM / 32, DDIM / 32, NEXP);   // W1 [E][D][F] -> W1t [E][F][D]
    moe_wtrans_kernel<<<g1, 256, 0, stream>>>(W1, W1t, DDIM, FDIM);
    dim3 g2(DDIM / 32, FDIM / 32, NEXP);   // W2 [E][F][D] -> W2t [E][D][F]
    moe_wtrans_kernel<<<g2, 256, 0, stream>>>(W2, W2t, FDIM, DDIM);
  }
  moe_ffn_kernel<<<T_TOK / MT, TPB, 0, stream>>>(hidden, W1t, b1, W2t, b2,
                                                 cwp, out);
}
